// ProtoNet_46351287059071
// MI455X (gfx1250) — matrix operand order repaired and checked by reading, not compiled, not run on hardware
//
#include <hip/hip_runtime.h>
#include <hip/hip_bf16.h>
#include <math.h>

// ---------------------------------------------------------------------------
// ProtoNet forward on MI455X (gfx1250, wave32, WMMA)
//   Z = [support; query] @ W + b            (bf16 WMMA, f32 accum, dbl-buffered)
//   proto = segment_mean(Z_support, labels) (VALU)
//   cscores = (Zq @ proto^T) / max(|q||p|, eps)
// ---------------------------------------------------------------------------

typedef __attribute__((ext_vector_type(16))) __bf16 v16bf;
typedef __attribute__((ext_vector_type(8)))  float  v8f;
typedef __attribute__((ext_vector_type(4)))  __bf16 bf16x4;
typedef __attribute__((ext_vector_type(2)))  __bf16 bf16x2;

#define K_DIM  49152
#define N_DIM  2048
#define M_TOT  3072
#define NSUP   1024
#define NQRY   2048
#define NWAY   64
#define BM     128
#define BN     128
#define BK     32

// ---------------------------------------------------------------------------
// Big GEMM: Z[M_TOT, N_DIM] = A[M_TOT, K_DIM] * W[K_DIM, N_DIM] + b
// A rows 0..1023 from support_images, 1024..3071 from query_images.
// Double-buffered LDS; tiles stored pre-permuted into WMMA fragment layouts
// so each lane reads its v16bf fragment as 32 contiguous bytes.
// ---------------------------------------------------------------------------
__global__ __launch_bounds__(256)
void protonet_gemm_wmma(const float* __restrict__ sup,
                        const float* __restrict__ qry,
                        const float* __restrict__ Wm,
                        const float* __restrict__ bias,
                        float* __restrict__ Z)
{
    // Fragment-ordered staging (ISA 7.12.2 16-bit layouts):
    //  lA: [msub(8)][lane(32)][j(16)]   A 16x32: lane half = K half-select
    //  lB: [nsub(8)][lane(32)][j(16)]   B 32x16: lanes 0-15 K=0..7,16..23; 16-31 K=8..15,24..31
    __shared__ __bf16 lA[2][BM * BK];   // 2 x 8 KB
    __shared__ __bf16 lB[2][BK * BN];   // 2 x 8 KB

    const int tid  = threadIdx.x;
    const int lane = tid & 31;
    const int wave = tid >> 5;       // 0..7
    const int wm   = wave >> 1;      // 0..3 : M wave coord (2 msubs)
    const int wn   = wave & 1;       // 0..1 : N wave coord (4 nsubs)
    const int m0   = blockIdx.y * BM;
    const int n0   = blockIdx.x * BN;

    // ---- A staging mapping: 32 rows x 8 thr/row per pass, 4 passes ----
    const int a_kq    = (tid & 7) << 2;                   // 0,4,...,28
    const int a_khalf = (a_kq & 15) >> 3;
    const int a_j0    = ((a_kq >> 4) << 3) + (a_kq & 7);
    const float* a_src[4];
    int a_off[4];
#pragma unroll
    for (int p = 0; p < 4; ++p) {
        const int rl = p * 32 + (tid >> 3);               // row in block tile
        const int gr = m0 + rl;
        a_src[p] = (gr < NSUP) ? (sup + (size_t)gr * K_DIM)
                               : (qry + (size_t)(gr - NSUP) * K_DIM);
        a_off[p] = (rl >> 4) * 512 + (a_khalf * 16 + (rl & 15)) * 16 + a_j0;
    }

    // ---- B staging mapping: 2 K-rows x 4 cols per thread per pass ----
    const int b_kp   = (tid >> 5) << 1;                   // 0,2,...,14
    const int b_nq   = (tid & 31) << 2;                   // 0,4,...,124
    const int b_nsub = b_nq >> 4;
    const int b_c16  = b_nq & 15;

    // ---- fragment read offsets ----
    int af_off[2], bf_off[4];
#pragma unroll
    for (int i = 0; i < 2; ++i) af_off[i] = (wm * 2 + i) * 512 + lane * 16;
#pragma unroll
    for (int j = 0; j < 4; ++j) bf_off[j] = (wn * 4 + j) * 512 + lane * 16;

    v8f acc[2][4];
#pragma unroll
    for (int i = 0; i < 2; ++i)
#pragma unroll
        for (int j = 0; j < 4; ++j) acc[i][j] = v8f{};

    // stage one K-chunk (global f32 -> bf16 -> fragment-ordered LDS)
    auto stage = [&](int k0, int buf) {
#pragma unroll
        for (int p = 0; p < 4; ++p) {
            float4 v = *(const float4*)(a_src[p] + k0 + a_kq);
            bf16x4 pv = { (__bf16)v.x, (__bf16)v.y, (__bf16)v.z, (__bf16)v.w };
            *(bf16x4*)&lA[buf][a_off[p]] = pv;
        }
#pragma unroll
        for (int p = 0; p < 2; ++p) {
            const int kk = ((b_kp >> 3) << 4) + p * 8 + (b_kp & 7); // khalf = p, j = b_kp
            const float* s0 = Wm + (size_t)(k0 + kk) * N_DIM + n0 + b_nq;
            float4 r0 = *(const float4*)(s0);
            float4 r1 = *(const float4*)(s0 + N_DIM);
            __bf16* dst = &lB[buf][b_nsub * 512 + (p * 16 + b_c16) * 16 + b_kp];
            *(bf16x2*)(dst)      = bf16x2{ (__bf16)r0.x, (__bf16)r1.x };
            *(bf16x2*)(dst + 16) = bf16x2{ (__bf16)r0.y, (__bf16)r1.y };
            *(bf16x2*)(dst + 32) = bf16x2{ (__bf16)r0.z, (__bf16)r1.z };
            *(bf16x2*)(dst + 48) = bf16x2{ (__bf16)r0.w, (__bf16)r1.w };
        }
    };

    constexpr int NK = K_DIM / BK;                        // 1536
    stage(0, 0);
    __syncthreads();

    for (int kt = 0; kt < NK; ++kt) {
        const int cur = kt & 1;
        if (kt + 1 < NK) stage((kt + 1) * BK, cur ^ 1);   // overlap with compute

        v16bf af[2], bfr[4];
#pragma unroll
        for (int i = 0; i < 2; ++i)
            af[i] = *(const v16bf*)&lA[cur][af_off[i]];
#pragma unroll
        for (int j = 0; j < 4; ++j)
            bfr[j] = *(const v16bf*)&lB[cur][bf_off[j]];
#pragma unroll
        for (int i = 0; i < 2; ++i)
#pragma unroll
            for (int j = 0; j < 4; ++j)
                acc[i][j] = __builtin_amdgcn_wmma_f32_16x16x32_bf16(
                    false, af[i], false, bfr[j],
                    (short)0, acc[i][j], false, false);

        __syncthreads();   // cur fully read + nxt fully written
    }

    // ---- epilogue: add bias, scatter per C/D lane map (ISA 7.12.2) ----
    const int colq = lane & 15;
    const int mh   = (lane >> 4) * 8;
#pragma unroll
    for (int i = 0; i < 2; ++i) {
        const int mbase = m0 + (wm * 2 + i) * 16 + mh;
#pragma unroll
        for (int j = 0; j < 4; ++j) {
            const int nc = n0 + (wn * 4 + j) * 16 + colq;
            const float bv = bias[nc];
#pragma unroll
            for (int v = 0; v < 8; ++v)
                Z[(size_t)(mbase + v) * N_DIM + nc] = acc[i][j][v] + bv;
        }
    }
}

// ---------------------------------------------------------------------------
// Per-class mean prototypes: proto[c, col] = mean over support rows w/ label c
// ---------------------------------------------------------------------------
__global__ __launch_bounds__(256)
void protonet_proto(const float* __restrict__ Z,
                    const int* __restrict__ labels,
                    float* __restrict__ proto)
{
    const int col = blockIdx.x * 256 + threadIdx.x;
    const int c   = blockIdx.y;
    float acc = 0.f, cnt = 0.f;
    for (int r = 0; r < NSUP; ++r) {
        if (labels[r] == c) {            // wave-uniform scalar test
            acc += Z[(size_t)r * N_DIM + col];
            cnt += 1.f;
        }
    }
    proto[(size_t)c * N_DIM + col] = acc / cnt;
}

// ---------------------------------------------------------------------------
// Row L2 norms: blocks 0..2047 -> query rows of Z; 2048..2111 -> proto rows
// ---------------------------------------------------------------------------
__global__ __launch_bounds__(256)
void protonet_norms(const float* __restrict__ Z,
                    const float* __restrict__ proto,
                    float* __restrict__ qn,
                    float* __restrict__ pn)
{
    __shared__ float red[256];
    const int tid = threadIdx.x;
    const int r   = blockIdx.x;
    const float* src;
    float* dst;
    int idx;
    if (r < NQRY) { src = Z + (size_t)(NSUP + r) * N_DIM; dst = qn; idx = r; }
    else          { src = proto + (size_t)(r - NQRY) * N_DIM; dst = pn; idx = r - NQRY; }

    float s = 0.f;
    for (int k = tid; k < N_DIM; k += 256) {
        float v = src[k];
        s += v * v;
    }
    red[tid] = s;
    __syncthreads();
    for (int off = 128; off > 0; off >>= 1) {
        if (tid < off) red[tid] += red[tid + off];
        __syncthreads();
    }
    if (tid == 0) dst[idx] = sqrtf(red[0]);
}

// ---------------------------------------------------------------------------
// Cosine scores: out[q, c] = dot(zq, proto_c) / max(qn*pn, eps)
// one thread per (q, c); proto (512 KB) is L2-resident.
// ---------------------------------------------------------------------------
__global__ __launch_bounds__(256)
void protonet_cosine(const float* __restrict__ Z,
                     const float* __restrict__ proto,
                     const float* __restrict__ qn,
                     const float* __restrict__ pn,
                     float* __restrict__ out)
{
    const int gid = blockIdx.x * 256 + threadIdx.x;
    const int q = gid >> 6;
    const int c = gid & 63;
    const float* zq = Z + (size_t)(NSUP + q) * N_DIM;
    const float* zp = proto + (size_t)c * N_DIM;
    float dot = 0.f;
    for (int k = 0; k < N_DIM; k += 4) {
        float4 a = *(const float4*)(zq + k);
        float4 b = *(const float4*)(zp + k);
        dot += a.x * b.x + a.y * b.y + a.z * b.z + a.w * b.w;
    }
    const float denom = fmaxf(qn[q] * pn[c], 1e-6f);
    out[(size_t)q * NWAY + c] = dot / denom;
}

// ---------------------------------------------------------------------------
extern "C" void kernel_launch(void* const* d_in, const int* in_sizes, int n_in,
                              void* d_out, int out_size, void* d_ws, size_t ws_size,
                              hipStream_t stream)
{
    const float* sup    = (const float*)d_in[0];   // [1024, 49152]
    const int*   labels = (const int*)  d_in[1];   // [1024]
    const float* qry    = (const float*)d_in[2];   // [2048, 49152]
    const float* Wm     = (const float*)d_in[3];   // [49152, 2048]
    const float* bias   = (const float*)d_in[4];   // [2048]
    float* out = (float*)d_out;                    // [2048, 64]

    // workspace layout (f32): Z[3072*2048] | proto[64*2048] | qn[2048] | pn[64]
    float* Z     = (float*)d_ws;
    float* proto = Z + (size_t)M_TOT * N_DIM;
    float* qn    = proto + (size_t)NWAY * N_DIM;
    float* pn    = qn + NQRY;

    dim3 g_gemm(N_DIM / BN, M_TOT / BM);           // (16, 24)
    protonet_gemm_wmma<<<g_gemm, 256, 0, stream>>>(sup, qry, Wm, bias, Z);

    protonet_proto<<<dim3(N_DIM / 256, NWAY), 256, 0, stream>>>(Z, labels, proto);

    protonet_norms<<<NQRY + NWAY, 256, 0, stream>>>(Z, proto, qn, pn);

    protonet_cosine<<<(NQRY * NWAY) / 256, 256, 0, stream>>>(Z, proto, qn, pn, out);
}
